// GINLayer_1769526526270
// MI455X (gfx1250) — matrix operand order repaired and checked by reading, not compiled, not run on hardware
//
#include <hip/hip_runtime.h>
#include <hip/hip_bf16.h>
#include <stdint.h>

// ---------------------------------------------------------------------------
// GIN layer for MI455X (gfx1250, wave32, WMMA bf16, TDM tile staging)
//   agg = segment_sum(vals * x[col], row);  h = (1+eps)*x + agg
//   h = relu(BN(h@W1+b1)); h = relu(BN(h@W2+b2))
// D = 256 channels fixed.
// ---------------------------------------------------------------------------

#define DCH 256

typedef float   v8f   __attribute__((ext_vector_type(8)));
typedef __bf16  v16bf __attribute__((ext_vector_type(16)));
typedef __bf16  v8bf  __attribute__((ext_vector_type(8)));

typedef unsigned int u32x4 __attribute__((ext_vector_type(4)));
typedef int          i32x4 __attribute__((ext_vector_type(4)));
typedef int          i32x8 __attribute__((ext_vector_type(8)));

#if defined(__has_builtin)
#  if __has_builtin(__builtin_amdgcn_tensor_load_to_lds)
#    define HAVE_TDM 1
#  endif
#endif

// ---------------- utility kernels ----------------

__global__ void k_zero_f32(float* __restrict__ p, int n) {
    int i = blockIdx.x * blockDim.x + threadIdx.x;
    int stride = gridDim.x * blockDim.x;
    for (; i < n; i += stride) p[i] = 0.0f;
}

// Wt[n][k] = bf16(W[k][n])  (transpose + convert; 256 blocks x 256 threads)
__global__ void k_convw(const float* __restrict__ W, __bf16* __restrict__ Wt) {
    int n = blockIdx.x;      // output column
    int k = threadIdx.x;     // input row
    Wt[(size_t)n * DCH + k] = (__bf16)W[(size_t)k * DCH + n];
}

// ---------------- SpMM: one wave32 per edge, 8 floats per lane ----------------

__global__ __launch_bounds__(256)
void k_spmm(const float* __restrict__ x,
            const int* __restrict__ erow, const int* __restrict__ ecol,
            const float* __restrict__ evals,
            float* __restrict__ agg, int E) {
    int gid  = blockIdx.x * blockDim.x + threadIdx.x;
    int e    = gid >> 5;
    int lane = gid & 31;
    if (e >= E) return;
    int   r = erow[e];
    int   c = ecol[e];
    float v = evals[e];
    const float4* xs = (const float4*)(x + (size_t)c * DCH);
    float4 a = xs[lane * 2 + 0];
    float4 b = xs[lane * 2 + 1];
    float* dst = agg + (size_t)r * DCH + lane * 8;
    atomicAdd(dst + 0, v * a.x);
    atomicAdd(dst + 1, v * a.y);
    atomicAdd(dst + 2, v * a.z);
    atomicAdd(dst + 3, v * a.w);
    atomicAdd(dst + 4, v * b.x);
    atomicAdd(dst + 5, v * b.y);
    atomicAdd(dst + 6, v * b.z);
    atomicAdd(dst + 7, v * b.w);
}

// h0_bf16 = bf16((1+eps)*x + agg)
__global__ void k_combine(const float* __restrict__ x, const float* __restrict__ agg,
                          const float* __restrict__ eps, __bf16* __restrict__ out, int total) {
    int i = blockIdx.x * blockDim.x + threadIdx.x;
    if (i >= total) return;
    float e = 1.0f + eps[0];
    out[i] = (__bf16)(e * x[i] + agg[i]);
}

// ---------------- TDM: stage a 64x256 bf16 tile into padded LDS ----------------
// D# describes: 2D tensor, data_size=2B, tile 256(x) x 64(y), tensor_dim1 =
// remaining rows (hardware zero-fills OOB reads -> tail-block padding for free),
// LDS padding of 4 DWORDs (16B) after every 128 DWORDs (512B = one row) ->
// reproduces the [64][264] bank-conflict-free layout in hardware.

#ifdef HAVE_TDM
__device__ __forceinline__ void tdm_load_tile(const __bf16* gptr, unsigned lds_byte_addr,
                                              int rows_here) {
    unsigned long long ga = (unsigned long long)(uintptr_t)gptr;
    u32x4 g0;
    g0[0] = 1u;                                             // count=1, user mode
    g0[1] = lds_byte_addr;                                  // lds_addr [63:32]
    g0[2] = (unsigned)(ga & 0xFFFFFFFFu);                   // global_addr lo
    g0[3] = (unsigned)((ga >> 32) & 0x1FFFFFFu) | (2u << 30); // global_addr hi | type=2
    i32x8 g1;
    g1[0] = (int)((1u << 16)      // data_size = 2 bytes
                | (1u << 20)      // pad_enable
                | (6u << 22)      // pad_interval: 128 DWORDs (512B) between pads
                | (3u << 25));    // pad_amount: 4 DWORDs (16B)
    g1[1] = (int)(256u << 16);                  // tensor_dim0 = 256 (bits 79:48)
    g1[2] = (int)((unsigned)rows_here << 16);   // tensor_dim1 = rows (bits 111:80)
    g1[3] = (int)(256u << 16);                  // tile_dim0 = 256 (bits 127:112)
    g1[4] = 64;                                 // tile_dim1 = 64 (bits 143:128)
    g1[5] = 256;                                // tensor_dim0_stride = 256
    g1[6] = 0;
    g1[7] = 0;
    i32x4 z4 = (i32x4){0, 0, 0, 0};
#if __clang_major__ >= 23
    i32x8 z8 = (i32x8){0, 0, 0, 0, 0, 0, 0, 0};
    __builtin_amdgcn_tensor_load_to_lds(g0, g1, z4, z4, z8, 0);
#else
    __builtin_amdgcn_tensor_load_to_lds(g0, g1, z4, z4, 0);
#endif
}
#endif

// ---------------- WMMA GEMM: Y[nrows,256] = H[nrows,256](bf16) @ W(256x256) + bias ----
// Wt is W transposed: Wt[n][k] bf16, so B-fragment columns are contiguous.
// Block: 256 threads = 8 waves laid out 4(M) x 2(N). Block tile: 64 x 256, K=256.

__global__ __launch_bounds__(256)
void k_gemm_wmma(const __bf16* __restrict__ H, const __bf16* __restrict__ Wt,
                 const float* __restrict__ bias, float* __restrict__ Y, int nrows) {
    // LDS A tile: 64 rows x 256 k, padded to 264 (16B pad/row) for conflict-free frags
    __shared__ __bf16 ldsA[64 * 264];

    const int m0 = blockIdx.x * 64;

#ifdef HAVE_TDM
    // ---- one wave issues the tensor DMA; hardware pads rows + zero-fills tail ----
    if ((threadIdx.x >> 5) == 0) {
        int rows_here = nrows - m0;
        if (rows_here > 64) rows_here = 64;
        unsigned laddr = (unsigned)(uintptr_t)(&ldsA[0]);   // low 32b of flat = LDS offset
        tdm_load_tile(H + (size_t)m0 * DCH, laddr, rows_here);
        __builtin_amdgcn_s_wait_tensorcnt(0);
    }
#else
    // ---- manual staging fallback: 64 x 256 bf16 = 32KB in 16B chunks ----
    {
        const uint4 zero = make_uint4(0u, 0u, 0u, 0u);
        #pragma unroll
        for (int i = 0; i < 8; ++i) {
            int chunk = i * 256 + threadIdx.x;    // 0..2047
            int row   = chunk >> 5;               // 0..63
            int kc    = chunk & 31;               // 16B chunk within row (8 bf16)
            int gr    = m0 + row;
            uint4 val = zero;
            if (gr < nrows)
                val = *(const uint4*)(H + (size_t)gr * DCH + kc * 8);
            *(uint4*)&ldsA[row * 264 + kc * 8] = val;
        }
    }
#endif
    __syncthreads();

    const int wave = threadIdx.x >> 5;
    const int lane = threadIdx.x & 31;
    const int mt   = wave & 3;     // M sub-tile (16 rows each)
    const int ng   = wave >> 2;    // N half (128 cols each)
    const int lhi  = lane >> 4;    // 0/1: lane group
    const int llo  = lane & 15;

    v8f acc[8];
    #pragma unroll
    for (int nt = 0; nt < 8; ++nt) acc[nt] = (v8f){0,0,0,0,0,0,0,0};

    const int mrow = mt * 16 + llo;         // row inside A tile for this lane

    #pragma unroll
    for (int kb = 0; kb < 8; ++kb) {
        // ---- A fragment: lane holds row mrow, k = kbase..kbase+15
        const int kbaseA = kb * 32 + (lhi << 4);
        const __bf16* pa = &ldsA[mrow * 264 + kbaseA];
        v8bf lo = *(const v8bf*)(pa);
        v8bf hi = *(const v8bf*)(pa + 8);
        v16bf afrag;
        #pragma unroll
        for (int j = 0; j < 8; ++j) { afrag[j] = lo[j]; afrag[j + 8] = hi[j]; }

        // ---- B fragments: lane holds column n, 16 consecutive k from kbaseB
        const int kbaseB = kb * 32 + (lhi << 4);
        #pragma unroll
        for (int nt = 0; nt < 8; ++nt) {
            int n = ng * 128 + nt * 16 + llo;
            v16bf bfrag = *(const v16bf*)(Wt + (size_t)n * DCH + kbaseB);
            acc[nt] = __builtin_amdgcn_wmma_f32_16x16x32_bf16(
                false, afrag, false, bfrag, (short)0, acc[nt], false, false);
        }
    }

    // ---- epilogue: D layout: lane -> N = llo, M = lhi*8 + r (r = vgpr index)
    const int mbase = m0 + mt * 16 + (lhi << 3);
    #pragma unroll
    for (int nt = 0; nt < 8; ++nt) {
        int n = ng * 128 + nt * 16 + llo;
        float bv = bias[n];
        #pragma unroll
        for (int r = 0; r < 8; ++r) {
            int m = mbase + r;
            if (m < nrows) Y[(size_t)m * DCH + n] = acc[nt][r] + bv;
        }
    }
}

// ---------------- per-channel stats: each block reduces 256 rows locally ----------------

__global__ __launch_bounds__(256)
void k_colstats(const float* __restrict__ Y, int nrows,
                float* __restrict__ sum, float* __restrict__ sumsq) {
    int c  = threadIdx.x;
    int r0 = blockIdx.x * 256;
    int r1 = min(r0 + 256, nrows);
    float s = 0.0f, q = 0.0f;
    for (int r = r0; r < r1; ++r) {
        float y = Y[(size_t)r * DCH + c];
        s += y;
        q += y * y;
    }
    atomicAdd(&sum[c], s);
    atomicAdd(&sumsq[c], q);
}

__global__ void k_finalize(const float* __restrict__ sum, const float* __restrict__ sumsq,
                           const float* __restrict__ g, const float* __restrict__ be,
                           float* __restrict__ scale, float* __restrict__ shift, float n) {
    int c = threadIdx.x;
    float mean = sum[c] / n;
    float var  = sumsq[c] / n - mean * mean;
    float sc   = g[c] * rsqrtf(var + 1e-5f);
    scale[c] = sc;
    shift[c] = be[c] - mean * sc;
}

__global__ void k_bnrelu_bf16(const float* __restrict__ Y, const float* __restrict__ scale,
                              const float* __restrict__ shift, __bf16* __restrict__ out, int total) {
    int i = blockIdx.x * blockDim.x + threadIdx.x;
    if (i >= total) return;
    int c = i & (DCH - 1);
    float h = fmaxf(Y[i] * scale[c] + shift[c], 0.0f);
    out[i] = (__bf16)h;
}

__global__ void k_bnrelu_f32(const float* __restrict__ Y, const float* __restrict__ scale,
                             const float* __restrict__ shift, float* __restrict__ out, int total) {
    int i = blockIdx.x * blockDim.x + threadIdx.x;
    if (i >= total) return;
    int c = i & (DCH - 1);
    out[i] = fmaxf(Y[i] * scale[c] + shift[c], 0.0f);
}

// ---------------- launch ----------------

extern "C" void kernel_launch(void* const* d_in, const int* in_sizes, int n_in,
                              void* d_out, int out_size, void* d_ws, size_t ws_size,
                              hipStream_t stream) {
    const float* x     = (const float*)d_in[0];
    const int*   erow  = (const int*)  d_in[1];
    const int*   ecol  = (const int*)  d_in[2];
    const float* evals = (const float*)d_in[3];
    const float* eps   = (const float*)d_in[4];
    const float* W1    = (const float*)d_in[5];
    const float* b1    = (const float*)d_in[6];
    const float* g1    = (const float*)d_in[7];
    const float* be1   = (const float*)d_in[8];
    const float* W2    = (const float*)d_in[9];
    const float* b2    = (const float*)d_in[10];
    const float* g2    = (const float*)d_in[11];
    const float* be2   = (const float*)d_in[12];

    const int N = in_sizes[0] / DCH;   // 100000
    const int E = in_sizes[1];         // 3200000
    const int total = N * DCH;

    // ---- workspace layout ----
    char* ws = (char*)d_ws;
    size_t off = 0;
    float* bufY = (float*)(ws + off);  off += (size_t)total * sizeof(float);        // agg / Y
    __bf16* hbf = (__bf16*)(ws + off); off += (size_t)total * sizeof(__bf16);       // bf16 activations
    __bf16* Wt1 = (__bf16*)(ws + off); off += (size_t)DCH * DCH * sizeof(__bf16);
    __bf16* Wt2 = (__bf16*)(ws + off); off += (size_t)DCH * DCH * sizeof(__bf16);
    float* sum   = (float*)(ws + off); off += DCH * sizeof(float);
    float* sumsq = (float*)(ws + off); off += DCH * sizeof(float);
    float* scale = (float*)(ws + off); off += DCH * sizeof(float);
    float* shift = (float*)(ws + off); off += DCH * sizeof(float);
    (void)ws_size; (void)n_in; (void)out_size;

    const int tpb = 256;
    const int gElem  = (total + tpb - 1) / tpb;
    const int gGemm  = (N + 63) / 64;
    const int gStats = (N + 255) / 256;

    // 1) zero agg
    k_zero_f32<<<4096, tpb, 0, stream>>>(bufY, total);
    // 2) convert + transpose weights to bf16
    k_convw<<<DCH, DCH, 0, stream>>>(W1, Wt1);
    k_convw<<<DCH, DCH, 0, stream>>>(W2, Wt2);
    // 3) SpMM scatter (one wave per edge)
    {
        long long threads = (long long)E * 32;
        int blocks = (int)((threads + tpb - 1) / tpb);
        k_spmm<<<blocks, tpb, 0, stream>>>(x, erow, ecol, evals, bufY, E);
    }
    // 4) combine + cast bf16
    k_combine<<<gElem, tpb, 0, stream>>>(x, bufY, eps, hbf, total);
    // 5) GEMM1 (overwrites bufY)
    k_gemm_wmma<<<gGemm, tpb, 0, stream>>>(hbf, Wt1, b1, bufY, N);
    // 6) BN1 stats
    k_zero_f32<<<1, 512, 0, stream>>>(sum, 2 * DCH);   // sum+sumsq contiguous
    k_colstats<<<gStats, tpb, 0, stream>>>(bufY, N, sum, sumsq);
    k_finalize<<<1, DCH, 0, stream>>>(sum, sumsq, g1, be1, scale, shift, (float)N);
    // 7) BN1+ReLU -> bf16
    k_bnrelu_bf16<<<gElem, tpb, 0, stream>>>(bufY, scale, shift, hbf, total);
    // 8) GEMM2
    k_gemm_wmma<<<gGemm, tpb, 0, stream>>>(hbf, Wt2, b2, bufY, N);
    // 9) BN2 stats
    k_zero_f32<<<1, 512, 0, stream>>>(sum, 2 * DCH);
    k_colstats<<<gStats, tpb, 0, stream>>>(bufY, N, sum, sumsq);
    k_finalize<<<1, DCH, 0, stream>>>(sum, sumsq, g2, be2, scale, shift, (float)N);
    // 10) BN2+ReLU -> f32 output
    k_bnrelu_f32<<<gElem, tpb, 0, stream>>>(bufY, scale, shift, (float*)d_out, total);
}
